// NewAttentionMLP_17858474017429
// MI455X (gfx1250) — compile-verified, matrix operand order checked by reading, not run on hardware
//
#include <hip/hip_runtime.h>
#include <hip/hip_bf16.h>

typedef __attribute__((ext_vector_type(16))) _Float16 v16h;
typedef __attribute__((ext_vector_type(8)))  float    v8f;

#define TPB 256
#define NWAVES 8

// ---------------------------------------------------------------------------
// Fragment loader for 16-bit 16x32 WMMA operand (A, or B loaded from Bt=[N x K]).
// Per ISA layout: lanes 0-15 hold K = {0..7, 16..23}, lanes 16-31 hold
// K = {8..15, 24..31}; each VGPR holds a (K, K+1) pair -> dword loads
// (compiler merges these into two global_load_b128 per fragment).
// `base` must already point at rowStart + k + khalf (khalf = (lane>=16)*8).
// ---------------------------------------------------------------------------
__device__ __forceinline__ v16h load_frag16(const _Float16* base) {
  union { v16h h; unsigned u[8]; } f;
#pragma unroll
  for (int j = 0; j < 4; ++j) {
    f.u[j]     = *(const unsigned*)(base + 2 * j);
    f.u[j + 4] = *(const unsigned*)(base + 16 + 2 * j);
  }
  return f.h;
}

// ---------------------------------------------------------------------------
// Generic WMMA GEMM:  C[M x N] = epilogue( A[M x K](f16) @ B + bias )
// B is provided transposed: Bt[N x K] f16 (K-contiguous rows).
// mode 0: relu(acc+bias) ; mode 1: res + relu(acc+bias) (res may == C) ;
// mode 2: acc+bias.
// Block = 8 waves arranged 4(M) x 2(N); each wave owns a 32x64 tile
// (2 A frags x 4 B frags -> 8 v_wmma per K-step, 12 b128 loads).
// grid = (N/128, M/128). Requires N%128==0, M%128==0, K%32==0.
// ---------------------------------------------------------------------------
__global__ __launch_bounds__(TPB)
void wmma_gemm_kernel(const _Float16* __restrict__ A,
                      const _Float16* __restrict__ Bt,
                      const float* __restrict__ bias,
                      const float* __restrict__ res,
                      float* __restrict__ C,
                      int M, int N, int K, int mode) {
  const int wave  = threadIdx.x >> 5;
  const int lane  = threadIdx.x & 31;
  const int m0    = blockIdx.y * 128 + (wave & 3) * 32;
  const int n0    = blockIdx.x * 128 + (wave >> 2) * 64;
  const int row   = lane & 15;
  const int khalf = (lane >> 4) << 3;   // 0 or 8

  const _Float16* Ap0 = A  + (size_t)(m0 +  0 + row) * K + khalf;
  const _Float16* Ap1 = A  + (size_t)(m0 + 16 + row) * K + khalf;
  const _Float16* Bp0 = Bt + (size_t)(n0 +  0 + row) * K + khalf;
  const _Float16* Bp1 = Bt + (size_t)(n0 + 16 + row) * K + khalf;
  const _Float16* Bp2 = Bt + (size_t)(n0 + 32 + row) * K + khalf;
  const _Float16* Bp3 = Bt + (size_t)(n0 + 48 + row) * K + khalf;

  v8f a00 = {}, a01 = {}, a02 = {}, a03 = {};
  v8f a10 = {}, a11 = {}, a12 = {}, a13 = {};

  for (int k = 0; k < K; k += 32) {
    __builtin_prefetch(Ap0 + k + 64, 0, 1);          // global_prefetch_b8
    v16h a0 = load_frag16(Ap0 + k);
    v16h a1 = load_frag16(Ap1 + k);
    v16h b0 = load_frag16(Bp0 + k);
    v16h b1 = load_frag16(Bp1 + k);
    v16h b2 = load_frag16(Bp2 + k);
    v16h b3 = load_frag16(Bp3 + k);
    a00 = __builtin_amdgcn_wmma_f32_16x16x32_f16(false, a0, false, b0, (short)0, a00, false, false);
    a01 = __builtin_amdgcn_wmma_f32_16x16x32_f16(false, a0, false, b1, (short)0, a01, false, false);
    a02 = __builtin_amdgcn_wmma_f32_16x16x32_f16(false, a0, false, b2, (short)0, a02, false, false);
    a03 = __builtin_amdgcn_wmma_f32_16x16x32_f16(false, a0, false, b3, (short)0, a03, false, false);
    a10 = __builtin_amdgcn_wmma_f32_16x16x32_f16(false, a1, false, b0, (short)0, a10, false, false);
    a11 = __builtin_amdgcn_wmma_f32_16x16x32_f16(false, a1, false, b1, (short)0, a11, false, false);
    a12 = __builtin_amdgcn_wmma_f32_16x16x32_f16(false, a1, false, b2, (short)0, a12, false, false);
    a13 = __builtin_amdgcn_wmma_f32_16x16x32_f16(false, a1, false, b3, (short)0, a13, false, false);
  }

  // D layout: lanes 0-15 -> col = lane, M = v ; lanes 16-31 -> col = lane-16, M = v+8
  const int col = lane & 15;
  const int mb  = (lane >> 4) << 3;
  v8f accs[2][4] = {{a00, a01, a02, a03}, {a10, a11, a12, a13}};
#pragma unroll
  for (int i = 0; i < 2; ++i) {
#pragma unroll
    for (int t = 0; t < 4; ++t) {
      const int n  = n0 + 16 * t + col;
      const float bv = bias[n];
#pragma unroll
      for (int v = 0; v < 8; ++v) {
        const size_t idx = (size_t)(m0 + 16 * i + mb + v) * N + n;
        float x = accs[i][t][v] + bv;
        if (mode < 2)  x = fmaxf(x, 0.f);
        if (mode == 1) x += res[idx];
        C[idx] = x;
      }
    }
  }
}

// ---------------------------------------------------------------------------
// Row LayerNorm, f32 -> f16, with zero padding to Cpad.  One block per row.
// ---------------------------------------------------------------------------
__global__ __launch_bounds__(TPB)
void ln_rows_kernel(const float* __restrict__ X,
                    const float* __restrict__ g,
                    const float* __restrict__ bta,
                    _Float16* __restrict__ Y,
                    int C, int Cpad) {
  const int r = blockIdx.x;
  const float* x = X + (size_t)r * C;
  _Float16*    y = Y + (size_t)r * Cpad;
  float s = 0.f, ss = 0.f;
  for (int i = threadIdx.x; i < C; i += TPB) { float v = x[i]; s += v; ss += v * v; }
  __shared__ float sh1[NWAVES], sh2[NWAVES];
#pragma unroll
  for (int off = 16; off > 0; off >>= 1) {
    s  += __shfl_down(s,  off, 32);
    ss += __shfl_down(ss, off, 32);
  }
  const int wave = threadIdx.x >> 5, lane = threadIdx.x & 31;
  if (lane == 0) { sh1[wave] = s; sh2[wave] = ss; }
  __syncthreads();
  if (threadIdx.x == 0) {
    float ts = 0.f, tss = 0.f;
    for (int i = 0; i < NWAVES; ++i) { ts += sh1[i]; tss += sh2[i]; }
    const float mean = ts / C;
    const float var  = tss / C - mean * mean;
    sh1[0] = mean;
    sh2[0] = rsqrtf(var + 1e-5f);
  }
  __syncthreads();
  const float mean = sh1[0], rstd = sh2[0];
  for (int i = threadIdx.x; i < C; i += TPB)
    y[i] = (_Float16)((x[i] - mean) * rstd * g[i] + bta[i]);
  for (int i = C + threadIdx.x; i < Cpad; i += TPB)
    y[i] = (_Float16)0.f;
}

// ---------------------------------------------------------------------------
// Attention input builder: for row r = b*OUT + o, gather the time column
// fc[(b*LN + j)*OUT + o] (j = 0..LN-1), append its mean, LayerNorm the
// (LN+1)-vector with per-position gamma/beta, emit f16 padded to Cpad.
// LN must equal TPB (256).
// ---------------------------------------------------------------------------
__global__ __launch_bounds__(TPB)
void attn_build_kernel(const float* __restrict__ fc,
                       const float* __restrict__ g,
                       const float* __restrict__ bta,
                       _Float16* __restrict__ Y,
                       int LN_, int OUT_, int Cpad) {
  const int r = blockIdx.x;
  const int b = r / OUT_;
  const int o = r % OUT_;
  const float v = fc[((size_t)b * LN_ + threadIdx.x) * OUT_ + o];
  float s = v, ss = v * v;
  __shared__ float sh1[NWAVES], sh2[NWAVES], stats[3];
#pragma unroll
  for (int off = 16; off > 0; off >>= 1) {
    s  += __shfl_down(s,  off, 32);
    ss += __shfl_down(ss, off, 32);
  }
  const int wave = threadIdx.x >> 5, lane = threadIdx.x & 31;
  if (lane == 0) { sh1[wave] = s; sh2[wave] = ss; }
  __syncthreads();
  if (threadIdx.x == 0) {
    float ts = 0.f, tss = 0.f;
    for (int i = 0; i < NWAVES; ++i) { ts += sh1[i]; tss += sh2[i]; }
    const float mval = ts / LN_;                   // appended mean element
    const float Csz  = (float)(LN_ + 1);
    const float mu   = (ts + mval) / Csz;
    const float var  = (tss + mval * mval) / Csz - mu * mu;
    stats[0] = mu;
    stats[1] = rsqrtf(var + 1e-5f);
    stats[2] = mval;
  }
  __syncthreads();
  const float mu = stats[0], rstd = stats[1];
  _Float16* y = Y + (size_t)r * Cpad;
  y[threadIdx.x] = (_Float16)((v - mu) * rstd * g[threadIdx.x] + bta[threadIdx.x]);
  if (threadIdx.x == 0)
    y[LN_] = (_Float16)((stats[2] - mu) * rstd * g[LN_] + bta[LN_]);
  for (int i = LN_ + 1 + threadIdx.x; i < Cpad; i += TPB)
    y[i] = (_Float16)0.f;
}

// ---------------------------------------------------------------------------
// In-place row softmax, C == TPB (256). One block per row.
// ---------------------------------------------------------------------------
__global__ __launch_bounds__(TPB)
void softmax_kernel(float* __restrict__ W, int C) {
  const size_t base = (size_t)blockIdx.x * C;
  float v = W[base + threadIdx.x];
  __shared__ float sh[NWAVES], red[1];
  float m = v;
#pragma unroll
  for (int off = 16; off > 0; off >>= 1) m = fmaxf(m, __shfl_down(m, off, 32));
  const int wave = threadIdx.x >> 5, lane = threadIdx.x & 31;
  if (lane == 0) sh[wave] = m;
  __syncthreads();
  if (threadIdx.x == 0) {
    float t = sh[0];
    for (int i = 1; i < NWAVES; ++i) t = fmaxf(t, sh[i]);
    red[0] = t;
  }
  __syncthreads();
  const float e = expf(v - red[0]);
  float s = e;
#pragma unroll
  for (int off = 16; off > 0; off >>= 1) s += __shfl_down(s, off, 32);
  if (lane == 0) sh[wave] = s;
  __syncthreads();
  if (threadIdx.x == 0) {
    float t = 0.f;
    for (int i = 0; i < NWAVES; ++i) t += sh[i];
    red[0] = t;
  }
  __syncthreads();
  W[base + threadIdx.x] = e / red[0];
}

// ---------------------------------------------------------------------------
// out[b*OUT + o] = sum_j fc[(b*LN + j)*OUT + o] * wsm[(b*OUT + o)*LN + j]
// ---------------------------------------------------------------------------
__global__ __launch_bounds__(TPB)
void final_kernel(const float* __restrict__ fc,
                  const float* __restrict__ wsm,
                  float* __restrict__ out,
                  int LN_, int OUT_) {
  const int r = blockIdx.x;
  const int b = r / OUT_;
  const int o = r % OUT_;
  float p = fc[((size_t)b * LN_ + threadIdx.x) * OUT_ + o] *
            wsm[(size_t)r * LN_ + threadIdx.x];
  __shared__ float sh[NWAVES];
#pragma unroll
  for (int off = 16; off > 0; off >>= 1) p += __shfl_down(p, off, 32);
  const int wave = threadIdx.x >> 5, lane = threadIdx.x & 31;
  if (lane == 0) sh[wave] = p;
  __syncthreads();
  if (threadIdx.x == 0) {
    float t = 0.f;
    for (int i = 0; i < NWAVES; ++i) t += sh[i];
    out[r] = t;
  }
}

// ---------------------------------------------------------------------------
// Weight convert + transpose:  W[K x N] f32  ->  Wt[N x Kpad] f16 (zero pad).
// ---------------------------------------------------------------------------
__global__ __launch_bounds__(TPB)
void convert_wt_kernel(const float* __restrict__ W,
                       _Float16* __restrict__ Wt,
                       int K, int N, int Kpad) {
  const long idx = (long)blockIdx.x * TPB + threadIdx.x;
  if (idx >= (long)N * Kpad) return;
  const int n = (int)(idx / Kpad);
  const int k = (int)(idx % Kpad);
  Wt[idx] = (k < K) ? (_Float16)W[(size_t)k * N + n] : (_Float16)0.f;
}

// ---------------------------------------------------------------------------
extern "C" void kernel_launch(void* const* d_in, const int* in_sizes, int n_in,
                              void* d_out, int out_size, void* d_ws, size_t ws_size,
                              hipStream_t stream) {
  (void)in_sizes; (void)n_in; (void)out_size; (void)ws_size;
  const int BS = 64, LNt = 256, FS = 512, H = 1024, OUT = 128, ATT = 256, L = 8;
  const int ROWS = BS * LNt;     // 16384
  const int AR   = BS * OUT;     // 8192 attention rows
  const int KA1  = 288;          // 257 padded to multiple of 32

  const float* x       = (const float*)d_in[0];
  const float* ln0_g   = (const float*)d_in[1];
  const float* ln0_b   = (const float*)d_in[2];
  const float* W0      = (const float*)d_in[3];
  const float* b0      = (const float*)d_in[4];
  const float* res_ln_g= (const float*)d_in[5];
  const float* res_ln_b= (const float*)d_in[6];
  const float* res_W   = (const float*)d_in[7];
  const float* res_b   = (const float*)d_in[8];
  const float* lnf_g   = (const float*)d_in[9];
  const float* lnf_b   = (const float*)d_in[10];
  const float* Wf      = (const float*)d_in[11];
  const float* bf      = (const float*)d_in[12];
  const float* a_ln1_g = (const float*)d_in[13];
  const float* a_ln1_b = (const float*)d_in[14];
  const float* Wa1     = (const float*)d_in[15];
  const float* ba1     = (const float*)d_in[16];
  const float* a_ln2_g = (const float*)d_in[17];
  const float* a_ln2_b = (const float*)d_in[18];
  const float* Wa2     = (const float*)d_in[19];
  const float* ba2     = (const float*)d_in[20];
  float* out           = (float*)d_out;

  // ---- workspace layout ----
  char* ws = (char*)d_ws;
  size_t off = 0;
  auto alloc = [&](size_t bytes) -> void* {
    off = (off + 255) & ~(size_t)255;
    void* p = ws + off;
    off += bytes;
    return p;
  };
  float*    h     = (float*)   alloc((size_t)ROWS * H   * 4);
  _Float16* xln   = (_Float16*)alloc((size_t)ROWS * H   * 2);
  float*    fc    = (float*)   alloc((size_t)ROWS * OUT * 4);
  _Float16* tln   = (_Float16*)alloc((size_t)AR * KA1   * 2);
  float*    abuf  = (float*)   alloc((size_t)AR * ATT   * 4);
  _Float16* aln   = (_Float16*)alloc((size_t)AR * ATT   * 2);
  float*    wbuf  = (float*)   alloc((size_t)AR * LNt   * 4);
  _Float16* W0t   = (_Float16*)alloc((size_t)H * FS     * 2);
  _Float16* resWt = (_Float16*)alloc((size_t)L * H * H  * 2);
  _Float16* Wft   = (_Float16*)alloc((size_t)OUT * H    * 2);
  _Float16* Wa1t  = (_Float16*)alloc((size_t)ATT * KA1  * 2);
  _Float16* Wa2t  = (_Float16*)alloc((size_t)LNt * ATT  * 2);

  auto cg = [](long total) { return dim3((unsigned)((total + TPB - 1) / TPB)); };

  // ---- weight conversion (f32 [KxN] -> f16 [N x Kpad], transposed) ----
  convert_wt_kernel<<<cg((long)H * FS), TPB, 0, stream>>>(W0, W0t, FS, H, FS);
  for (int i = 0; i < L; ++i)
    convert_wt_kernel<<<cg((long)H * H), TPB, 0, stream>>>(
        res_W + (size_t)i * H * H, resWt + (size_t)i * H * H, H, H, H);
  convert_wt_kernel<<<cg((long)OUT * H), TPB, 0, stream>>>(Wf, Wft, H, OUT, H);
  convert_wt_kernel<<<cg((long)ATT * KA1), TPB, 0, stream>>>(Wa1, Wa1t, LNt + 1, ATT, KA1);
  convert_wt_kernel<<<cg((long)LNt * ATT), TPB, 0, stream>>>(Wa2, Wa2t, ATT, LNt, ATT);

  // ---- stem: h = relu(LN(x) @ W0 + b0) ----
  ln_rows_kernel<<<ROWS, TPB, 0, stream>>>(x, ln0_g, ln0_b, xln, FS, FS);
  wmma_gemm_kernel<<<dim3(H / 128, ROWS / 128), TPB, 0, stream>>>(
      xln, W0t, b0, nullptr, h, ROWS, H, FS, 0);

  // ---- residual stack: h += relu(LN(h) @ Wi + bi) ----
  for (int i = 0; i < L; ++i) {
    ln_rows_kernel<<<ROWS, TPB, 0, stream>>>(
        h, res_ln_g + (size_t)i * H, res_ln_b + (size_t)i * H, xln, H, H);
    wmma_gemm_kernel<<<dim3(H / 128, ROWS / 128), TPB, 0, stream>>>(
        xln, resWt + (size_t)i * H * H, res_b + (size_t)i * H, h, h, ROWS, H, H, 1);
  }

  // ---- head: fc = LN(h) @ Wf + bf ----
  ln_rows_kernel<<<ROWS, TPB, 0, stream>>>(h, lnf_g, lnf_b, xln, H, H);
  wmma_gemm_kernel<<<dim3(OUT / 128, ROWS / 128), TPB, 0, stream>>>(
      xln, Wft, bf, nullptr, fc, ROWS, OUT, H, 2);

  // ---- attention MLP over time axis ----
  attn_build_kernel<<<AR, TPB, 0, stream>>>(fc, a_ln1_g, a_ln1_b, tln, LNt, OUT, KA1);
  wmma_gemm_kernel<<<dim3(ATT / 128, AR / 128), TPB, 0, stream>>>(
      tln, Wa1t, ba1, nullptr, abuf, AR, ATT, KA1, 0);
  ln_rows_kernel<<<AR, TPB, 0, stream>>>(abuf, a_ln2_g, a_ln2_b, aln, ATT, ATT);
  wmma_gemm_kernel<<<dim3(LNt / 128, AR / 128), TPB, 0, stream>>>(
      aln, Wa2t, ba2, nullptr, wbuf, AR, LNt, ATT, 2);
  softmax_kernel<<<AR, TPB, 0, stream>>>(wbuf, LNt);

  // ---- weighted sum over time ----
  final_kernel<<<AR, TPB, 0, stream>>>(fc, wbuf, out, LNt, OUT);
}
